// TernaryTransformerBlock_49160195670158
// MI455X (gfx1250) — matrix operand order repaired and checked by reading, not compiled, not run on hardware
//
#include <hip/hip_runtime.h>

// ---------------- problem constants (from reference setup) ----------------
#define BB   8
#define SS   2048
#define FF   2048
#define MTOT (BB*SS)          // 16384 rows of the GEMM

typedef __attribute__((ext_vector_type(16))) __bf16 v16bf;
typedef __attribute__((ext_vector_type(8)))  float  v8f;

union Frag { v16bf v; uint4 q[2]; };

__device__ __forceinline__ unsigned short f2bf(float f) {
    unsigned int u = __float_as_uint(f);
    unsigned int r = u + 0x7FFFu + ((u >> 16) & 1u);   // round-to-nearest-even
    return (unsigned short)(r >> 16);
}
__device__ __forceinline__ unsigned int pk2(float lo, float hi) {
    return (unsigned int)f2bf(lo) | ((unsigned int)f2bf(hi) << 16);
}

__device__ __forceinline__ float blockReduce256(float v, float* red) {
    int t = threadIdx.x;
    red[t] = v; __syncthreads();
    for (int off = 128; off > 0; off >>= 1) {
        if (t < off) red[t] += red[t + off];
        __syncthreads();
    }
    float r = red[0]; __syncthreads();
    return r;
}

// ---------------- 1) scale = mean(|W|), deterministic 2-pass ----------------
__global__ __launch_bounds__(256) void absw_partial_kernel(
    const float* __restrict__ W, float* __restrict__ part, int N) {
    __shared__ float red[256];
    float s = 0.f;
    for (int i = blockIdx.x * 256 + threadIdx.x; i < N; i += gridDim.x * 256)
        s += fabsf(W[i]);
    s = blockReduce256(s, red);
    if (threadIdx.x == 0) part[blockIdx.x] = s;
}

__global__ __launch_bounds__(256) void absw_final_kernel(
    const float* __restrict__ part, float* __restrict__ scaleOut, int nPart, float invCount) {
    __shared__ float red[256];
    float s = 0.f;
    for (int i = threadIdx.x; i < nPart; i += 256) s += part[i];
    s = blockReduce256(s, red);
    if (threadIdx.x == 0) *scaleOut = s * invCount;
}

// ------- 2) ternary quantize W (no scale) + transpose -> bf16 Wqt[g][f] -------
__global__ __launch_bounds__(256) void quant_transpose_kernel(
    const float* __restrict__ W, const float* __restrict__ scaleP,
    unsigned short* __restrict__ Wqt) {
    __shared__ float tile[32][33];
    float invS = 1.0f / (*scaleP + 1e-8f);
    int gBase = blockIdx.x * 32, fBase = blockIdx.y * 32;
    int tx = threadIdx.x, ty = threadIdx.y;           // 32 x 8
    #pragma unroll
    for (int i = 0; i < 4; ++i) {
        int f = ty + i * 8;
        tile[f][tx] = W[(size_t)(fBase + f) * FF + gBase + tx];
    }
    __syncthreads();
    #pragma unroll
    for (int i = 0; i < 4; ++i) {
        int g = ty + i * 8;
        float w = tile[tx][g];                         // W[fBase+tx][gBase+g]
        float q = rintf(w * invS);
        q = fminf(1.0f, fmaxf(-1.0f, q));
        Wqt[(size_t)(gBase + g) * FF + fBase + tx] = f2bf(q);
    }
}

// --------- 2b) pre-convert x (f32) -> bf16 once, streaming, RNE ---------
__global__ __launch_bounds__(256) void x_to_bf16_kernel(
    const float* __restrict__ x, unsigned short* __restrict__ Xb) {
    size_t gid = (size_t)blockIdx.x * 256 + threadIdx.x;   // 8 floats each
    const float4* src = (const float4*)x + gid * 2;
    float4 f0 = src[0], f1 = src[1];
    ((uint4*)Xb)[gid] = make_uint4(pk2(f0.x, f0.y), pk2(f0.z, f0.w),
                                   pk2(f1.x, f1.y), pk2(f1.z, f1.w));
}

// ---------------- 3) la_raw[row] = x[row,:] . w_left + b_left ----------------
__global__ __launch_bounds__(256) void la_raw_kernel(
    const float* __restrict__ x, const float* __restrict__ wl,
    const float* __restrict__ bl, float* __restrict__ la) {
    __shared__ float red[256];
    int row = blockIdx.x, t = threadIdx.x;
    const float* xr = x + (size_t)row * FF;
    float s = 0.f;
    #pragma unroll
    for (int i = 0; i < FF / 256; ++i) {
        int f = t + i * 256;
        s += xr[f] * wl[f];
    }
    s = blockReduce256(s, red);
    if (t == 0) la[row] = s + bl[0];
}

// ---------- 4) ra_raw[b,f] = sum_s x[b,s,f] * w_right[s] + b_right ----------
__global__ __launch_bounds__(256) void ra_raw_kernel(
    const float* __restrict__ x, const float* __restrict__ wr,
    const float* __restrict__ br, float* __restrict__ ra) {
    int gid = blockIdx.x * 256 + threadIdx.x;          // 0..B*F-1
    int b = gid >> 11, f = gid & (FF - 1);
    const float* xb = x + (size_t)b * SS * FF + f;
    float s = 0.f;
    for (int sidx = 0; sidx < SS; ++sidx)
        s += xb[(size_t)sidx * FF] * wr[sidx];
    ra[gid] = s + br[0];
}

// ------ 5) la: LayerNorm over S (per b) -> ReLU -> r/(r+1e-6), in place ------
__global__ __launch_bounds__(256) void ln_la_kernel(float* __restrict__ la) {
    __shared__ float red[256];
    int b = blockIdx.x, t = threadIdx.x;
    float* p = la + (size_t)b * SS;
    float vals[SS / 256];
    float s = 0.f;
    #pragma unroll
    for (int i = 0; i < SS / 256; ++i) { vals[i] = p[t + i * 256]; s += vals[i]; }
    float mu = blockReduce256(s, red) * (1.0f / SS);
    float vs = 0.f;
    #pragma unroll
    for (int i = 0; i < SS / 256; ++i) { float d = vals[i] - mu; vs += d * d; }
    float var = blockReduce256(vs, red) * (1.0f / SS);
    float inv = rsqrtf(var + 1e-5f);
    #pragma unroll
    for (int i = 0; i < SS / 256; ++i) {
        float r = fmaxf((vals[i] - mu) * inv, 0.0f);
        p[t + i * 256] = r / (r + 1e-6f);
    }
}

// -- 6) ra: LayerNorm over F (per b) -> ReLU -> r / (sum_F r + 1e-6), in place --
__global__ __launch_bounds__(256) void ln_ra_kernel(float* __restrict__ ra) {
    __shared__ float red[256];
    int b = blockIdx.x, t = threadIdx.x;
    float* p = ra + (size_t)b * FF;
    float vals[FF / 256];
    float s = 0.f;
    #pragma unroll
    for (int i = 0; i < FF / 256; ++i) { vals[i] = p[t + i * 256]; s += vals[i]; }
    float mu = blockReduce256(s, red) * (1.0f / FF);
    float vs = 0.f;
    #pragma unroll
    for (int i = 0; i < FF / 256; ++i) { float d = vals[i] - mu; vs += d * d; }
    float var = blockReduce256(vs, red) * (1.0f / FF);
    float inv = rsqrtf(var + 1e-5f);
    float rs = 0.f;
    float rv[FF / 256];
    #pragma unroll
    for (int i = 0; i < FF / 256; ++i) {
        rv[i] = fmaxf((vals[i] - mu) * inv, 0.0f);
        rs += rv[i];
    }
    float tot = blockReduce256(rs, red);
    float invTot = 1.0f / (tot + 1e-6f);
    #pragma unroll
    for (int i = 0; i < FF / 256; ++i) p[t + i * 256] = rv[i] * invTot;
}

// ---------------- 7) WMMA bf16 GEMM + fused epilogue ----------------
// block tile 128x128, 8 waves -> 32x64 per wave (2x4 of 16x16), K staged 32/iter.
// Tiles are brought in with CDNA5 async global->LDS DMA (ASYNCcnt), double-buffered.
// LDS pad stride 40 halves (80B row): conflict-free banks, 16B-aligned ds_load_b128.
#define LDA 40
#define KSTEP 32
#define NK (FF / KSTEP)

__global__ __launch_bounds__(256) void gemm_epilogue_kernel(
    const float* __restrict__ x, const unsigned short* __restrict__ Xb,
    const unsigned short* __restrict__ Wqt,
    const float* __restrict__ bv, const float* __restrict__ la,
    const float* __restrict__ ra, const float* __restrict__ scaleP,
    const float* __restrict__ alphaP, float* __restrict__ out) {

    __shared__ unsigned short As[2][128 * LDA];
    __shared__ unsigned short Bs[2][128 * LDA];

    const int t = threadIdx.x;
    const int lane = t & 31;
    const int hi = lane >> 4;          // lane half
    const int lr = lane & 15;
    const int wave = t >> 5;           // 0..7
    const int waveM = wave & 3;        // 4 waves along M (32 rows each)
    const int waveN = wave >> 2;       // 2 waves along N (64 cols each)

    const int rowBase = blockIdx.y * 128;   // over M = B*S
    const int colBase = blockIdx.x * 128;   // over N = F

    v8f acc[2][4];
    #pragma unroll
    for (int mt = 0; mt < 2; ++mt)
        #pragma unroll
        for (int nt = 0; nt < 4; ++nt) acc[mt][nt] = 0;

    const int lrow = t >> 1;           // 0..127 (tile row this thread copies)
    const int half = t & 1;            // which 32B half of the 64B row

    // Async DMA of one K-stage: A tile (bf16 x) and B tile (ternary bf16 W^T).
    // GVS addressing: SGPR64 base + VGPR32 byte offset; the `offset:` immediate
    // is added to BOTH the global and the LDS address (ISA 15.18.3).
    auto load_tile_async = [&](int buf, int kt) {
        const int k0 = kt * KSTEP;
        unsigned dstA = (unsigned)(size_t)&As[buf][lrow * LDA + half * 16];
        unsigned offA = (unsigned)(((rowBase + lrow) * FF + k0 + half * 16) * 2);
        unsigned dstB = (unsigned)(size_t)&Bs[buf][lrow * LDA + half * 16];
        unsigned offB = (unsigned)(((colBase + lrow) * FF + k0 + half * 16) * 2);
        asm volatile(
            "global_load_async_to_lds_b128 %0, %1, %4\n\t"
            "global_load_async_to_lds_b128 %0, %1, %4 offset:16\n\t"
            "global_load_async_to_lds_b128 %2, %3, %5\n\t"
            "global_load_async_to_lds_b128 %2, %3, %5 offset:16"
            :
            : "v"(dstA), "v"(offA), "v"(dstB), "v"(offB), "s"(Xb), "s"(Wqt)
            : "memory");
    };

    load_tile_async(0, 0);

    for (int kt = 0; kt < NK; ++kt) {
        // drain own async DMA, then make it visible to all waves
        asm volatile("s_wait_asynccnt 0" ::: "memory");
        __syncthreads();
        if (kt + 1 < NK) load_tile_async((kt + 1) & 1, kt + 1);

        const int cur = kt & 1;
        const unsigned short* Ab = As[cur];
        const unsigned short* Bb = Bs[cur];

        Frag fa[2], fb[4];
        #pragma unroll
        for (int mt = 0; mt < 2; ++mt) {
            // A 16-bit 16x32: lanes 0-15 K=0-7 & 16-23, lanes 16-31 K=8-15 & 24-31
            const unsigned short* ab = Ab + (waveM * 32 + mt * 16 + lr) * LDA;
            fa[mt].q[0] = *(const uint4*)(ab + hi * 8);
            fa[mt].q[1] = *(const uint4*)(ab + 16 + hi * 8);
        }
        #pragma unroll
        for (int nt = 0; nt < 4; ++nt) {
            // B 16-bit 32x16: lanes 0-15 K=0-7 & 16-23, lanes 16-31 K=8-15 & 24-31 (N-major tile)
            const unsigned short* bb = Bb + (waveN * 64 + nt * 16 + lr) * LDA + hi * 8;
            fb[nt].q[0] = *(const uint4*)(bb + 0);
            fb[nt].q[1] = *(const uint4*)(bb + 16);
        }
        #pragma unroll
        for (int mt = 0; mt < 2; ++mt)
            #pragma unroll
            for (int nt = 0; nt < 4; ++nt)
                acc[mt][nt] = __builtin_amdgcn_wmma_f32_16x16x32_bf16(
                    false, fa[mt].v, false, fb[nt].v,
                    (short)0, acc[mt][nt], false, false);
    }

    // fused epilogue: out = (1-a)*x + a * la * (scale*acc + b_v) * ra
    const float scl = *scaleP;
    const float a = *alphaP;
    const float oma = 1.0f - a;

    #pragma unroll
    for (int mt = 0; mt < 2; ++mt) {
        const int rowB = rowBase + waveM * 32 + mt * 16;
        #pragma unroll
        for (int nt = 0; nt < 4; ++nt) {
            const int col = colBase + waveN * 64 + nt * 16 + lr;
            const float bvc = bv[col];
            #pragma unroll
            for (int r = 0; r < 8; ++r) {
                const int row = rowB + r + hi * 8;       // C/D layout: lanes 0-15 M=r, 16-31 M=8+r
                const int b = row >> 11;                  // row / S
                const size_t idx = (size_t)row * FF + col;
                const float vv = scl * acc[mt][nt][r] + bvc;
                out[idx] = oma * x[idx] + a * la[row] * vv * ra[(b << 11) + col];
            }
        }
    }
}

// ---------------- host-side launcher ----------------
extern "C" void kernel_launch(void* const* d_in, const int* in_sizes, int n_in,
                              void* d_out, int out_size, void* d_ws, size_t ws_size,
                              hipStream_t stream) {
    const float* x   = (const float*)d_in[0];
    const float* Wv  = (const float*)d_in[3];
    const float* bv  = (const float*)d_in[4];
    const float* wl  = (const float*)d_in[5];
    const float* bl  = (const float*)d_in[6];
    const float* wr  = (const float*)d_in[7];
    const float* br  = (const float*)d_in[8];
    const float* alp = (const float*)d_in[9];
    float* out = (float*)d_out;

    // workspace layout (floats, then bf16 blobs)
    float* wsf    = (float*)d_ws;
    float* scaleP = wsf;                    // [1]
    float* part   = wsf + 16;               // [1024]
    float* la     = wsf + 2048;             // [B*S]
    float* ra     = wsf + 2048 + MTOT;      // [B*F]
    unsigned short* Wqt = (unsigned short*)(wsf + 2048 + MTOT + BB * FF); // [F*F] bf16 (8 MB)
    unsigned short* Xb  = Wqt + (size_t)FF * FF;                          // [M*F] bf16 (64 MB)

    const int NW = FF * FF;

    absw_partial_kernel<<<1024, 256, 0, stream>>>(Wv, part, NW);
    absw_final_kernel<<<1, 256, 0, stream>>>(part, scaleP, 1024, 1.0f / (float)NW);

    quant_transpose_kernel<<<dim3(FF / 32, FF / 32), dim3(32, 8), 0, stream>>>(Wv, scaleP, Wqt);

    x_to_bf16_kernel<<<(MTOT * (size_t)FF / 8) / 256, 256, 0, stream>>>(x, Xb);

    la_raw_kernel<<<MTOT, 256, 0, stream>>>(x, wl, bl, la);
    ln_la_kernel<<<BB, 256, 0, stream>>>(la);

    ra_raw_kernel<<<(BB * FF) / 256, 256, 0, stream>>>(x, wr, br, ra);
    ln_ra_kernel<<<BB, 256, 0, stream>>>(ra);

    gemm_epilogue_kernel<<<dim3(FF / 128, MTOT / 128), 256, 0, stream>>>(
        x, Xb, Wqt, bv, la, ra, scaleP, alp, out);
}
